// PositionalScaledDotProductAttention_77120432767769
// MI455X (gfx1250) — matrix operand order repaired and checked by reading, not compiled, not run on hardware
//
#include <hip/hip_runtime.h>
#include <hip/hip_bf16.h>

// MI455X / gfx1250, wave32. v_wmma_f32_16x16x32_f16 for all three contractions.
// rpr (512 MB, the roofline-dominant tensor) is read exactly once, streamed
// via the CDNA5 async global->LDS path (ASYNCcnt) with double buffering.

typedef _Float16 v4h  __attribute__((ext_vector_type(4)));
typedef _Float16 v8h  __attribute__((ext_vector_type(8)));
typedef _Float16 v16h __attribute__((ext_vector_type(16)));
typedef float    v8f  __attribute__((ext_vector_type(8)));

#define B_    2
#define H_    8
#define S_    1024
#define D_    64
#define INV_T 0.125f   // 1/sqrt(64)

static __device__ __forceinline__ v16h cat16(v8h lo, v8h hi) {
  return __builtin_shufflevector(lo, hi, 0,1,2,3,4,5,6,7,8,9,10,11,12,13,14,15);
}

// Convert 16 contiguous f32 (LDS) into a v16h WMMA operand.
static __device__ __forceinline__ v16h cvt16f(const float* p) {
  v16h r;
#pragma unroll
  for (int e = 0; e < 16; ++e) r[e] = (_Float16)p[e];
  return r;
}

// CDNA5 async copy: 16 bytes global -> LDS per lane, tracked by ASYNCcnt.
// lds = workgroup-relative LDS byte offset (low 32 bits of the flat address),
// gaddr = 64-bit global address (GV mode).
static __device__ __forceinline__ void async_b128(unsigned int lds,
                                                  unsigned long long gaddr) {
  asm volatile("global_load_async_to_lds_b128 %0, %1, off"
               :: "v"(lds), "v"(gaddr)
               : "memory");
}

static __device__ __forceinline__ unsigned int lds_off(const void* p) {
  return (unsigned int)(unsigned long long)(uintptr_t)p;  // addr[31:0] == LDS offset
}

// -----------------------------------------------------------------------------
// Kernel 1: attn2[b,h,i,j] = sum_d qs[b,h,i,d] * rpr[b,i,j,d]
// One wave per (b,i). A = 8 heads of qs (rows 8..15 zero), B = rpr tile^T.
// rpr tiles stream via double-buffered async global->LDS copies.
// Writes attn2 into the attn region of d_out (scratch, overwritten by kernel 2).
// -----------------------------------------------------------------------------
__global__ __launch_bounds__(32)
void attn2_kernel(const float* __restrict__ q, const float* __restrict__ rpr,
                  float* __restrict__ attn) {
  __shared__ __attribute__((aligned(32))) _Float16 qh[16][64];
  __shared__ __attribute__((aligned(16))) float    rtf[2][16][64];  // 8 KB, dbl-buffered

  const int wg   = blockIdx.x;         // 0 .. B*S-1
  const int b    = wg >> 10;           // wg / S
  const int i    = wg & (S_ - 1);
  const int lane = threadIdx.x;
  const int nm   = lane & 15;
  const int hiH  = (lane >> 4) & 1;    // 0: lanes 0-15, 1: lanes 16-31

  const float* rbase = rpr + (((size_t)b * S_ + i) * S_) * D_;  // rpr[b,i,0,0]

  // Issue async copy of the first rpr tile (16x64 f32 = 4 KB, contiguous).
  {
    const unsigned int l0 = lds_off(&rtf[0][0][0]);
    const unsigned long long g0 = (unsigned long long)(uintptr_t)rbase;
#pragma unroll
    for (int u = 0; u < 8; ++u) {
      const unsigned int off = (unsigned int)(u * 32 + lane) * 16u;
      async_b128(l0 + off, g0 + off);
    }
  }

  // Stage scaled q for all 8 heads; zero pad rows 8..15.
  {
    const int h     = lane >> 2;             // 0..7
    const int dbase = (lane & 3) * 16;
    const float4* qp =
        (const float4*)(q + (((size_t)b * H_ + h) * S_ + i) * D_ + dbase);
#pragma unroll
    for (int u = 0; u < 4; ++u) {
      float4 f = qp[u];
      *(v4h*)&qh[h][dbase + 4 * u] =
          (v4h){(_Float16)(f.x * INV_T), (_Float16)(f.y * INV_T),
                (_Float16)(f.z * INV_T), (_Float16)(f.w * INV_T)};
      *(v4h*)&qh[h + 8][dbase + 4 * u] = (v4h){(_Float16)0, (_Float16)0,
                                               (_Float16)0, (_Float16)0};
    }
  }
  __syncthreads();

  // A operands (constant over the j loop). Contiguous k = 16*half + s, matching B.
  v16h A0, A1;
  {
    const int koff = hiH * 16;
    A0 = cat16(*(const v8h*)&qh[nm][0  + koff], *(const v8h*)&qh[nm][8  + koff]);
    A1 = cat16(*(const v8h*)&qh[nm][32 + koff], *(const v8h*)&qh[nm][40 + koff]);
  }

  float* abase = attn + (size_t)b * H_ * S_ * S_ + (size_t)i * S_;  // attn[b,0,i,0]

  for (int j0 = 0; j0 < S_; j0 += 16) {
    const int p = (j0 >> 4) & 1;

    if (j0 + 16 < S_) {
      // Prefetch next tile into the other buffer, then wait for current tile.
      const unsigned int l1 = lds_off(&rtf[p ^ 1][0][0]);
      const unsigned long long g1 =
          (unsigned long long)(uintptr_t)(rbase + (size_t)(j0 + 16) * D_);
#pragma unroll
      for (int u = 0; u < 8; ++u) {
        const unsigned int off = (unsigned int)(u * 32 + lane) * 16u;
        async_b128(l1 + off, g1 + off);
      }
      // Async loads complete in order: <=8 outstanding => current tile landed.
      asm volatile("s_wait_asynccnt 8" ::: "memory");
    } else {
      asm volatile("s_wait_asynccnt 0" ::: "memory");
    }

    const int koffB = hiH * 16;                 // contiguous k = 16*half + s, matching A
    v16h B0 = cvt16f(&rtf[p][nm][0  + koffB]);
    v16h B1 = cvt16f(&rtf[p][nm][32 + koffB]);

    v8f acc = {0.f, 0.f, 0.f, 0.f, 0.f, 0.f, 0.f, 0.f};
    acc = __builtin_amdgcn_wmma_f32_16x16x32_f16(false, A0, false, B0, (short)0, acc, false, false);
    acc = __builtin_amdgcn_wmma_f32_16x16x32_f16(false, A1, false, B1, (short)0, acc, false, false);

    // D rows 0..7 (lanes 0..15) == heads 0..7
    if (lane < 16) {
#pragma unroll
      for (int r = 0; r < 8; ++r)
        abase[(size_t)r * S_ * S_ + j0 + nm] = acc[r];
    }
  }
}

// -----------------------------------------------------------------------------
// Kernel 2: per (b,h, 16-row i-tile): QK^T (WMMA) + attn2 + mask -> logits in
// LDS -> softmax -> write attn -> P*V (WMMA) -> write output. 4 waves.
// K tiles stream via wave-private async global->LDS copies.
// -----------------------------------------------------------------------------
__global__ __launch_bounds__(128)
void attn_fused_kernel(const float* __restrict__ q, const float* __restrict__ k,
                       const float* __restrict__ v, const int* __restrict__ mask,
                       float* __restrict__ attn, float* __restrict__ out) {
  __shared__ __attribute__((aligned(32))) float    Lg[16][S_];        // 64 KB logits/probs
  __shared__ __attribute__((aligned(32))) _Float16 qt[16][64];        // 2 KB
  __shared__ __attribute__((aligned(16))) float    kstf[4][16][64];   // 16 KB, per-wave
  __shared__ __attribute__((aligned(32))) _Float16 vt[64][32];        // 4 KB, transposed V chunk
  __shared__ float red[16][8];
  __shared__ float rowmax[16], rowsum[16];

  const int wg    = blockIdx.x;        // 0 .. B*H*(S/16)-1
  const int i0    = (wg & 63) * 16;
  const int bh    = wg >> 6;           // b*H + h
  const int b     = bh >> 3;
  const int tid   = threadIdx.x;
  const int wave  = tid >> 5;
  const int lane  = tid & 31;
  const int nm    = lane & 15;
  const int hiH   = (lane >> 4) & 1;

  // Stage scaled q tile (16 x 64)
  {
    const int row   = tid >> 3;
    const int dbase = (tid & 7) * 8;
    const float4* qp = (const float4*)(q + (((size_t)bh) * S_ + i0 + row) * D_ + dbase);
    float4 f0 = qp[0], f1 = qp[1];
    *(v4h*)&qt[row][dbase] =
        (v4h){(_Float16)(f0.x * INV_T), (_Float16)(f0.y * INV_T),
              (_Float16)(f0.z * INV_T), (_Float16)(f0.w * INV_T)};
    *(v4h*)&qt[row][dbase + 4] =
        (v4h){(_Float16)(f1.x * INV_T), (_Float16)(f1.y * INV_T),
              (_Float16)(f1.z * INV_T), (_Float16)(f1.w * INV_T)};
  }
  __syncthreads();

  v16h A0, A1;
  {
    const int koff = hiH * 16;
    A0 = cat16(*(const v8h*)&qt[nm][0  + koff], *(const v8h*)&qt[nm][8  + koff]);
    A1 = cat16(*(const v8h*)&qt[nm][32 + koff], *(const v8h*)&qt[nm][40 + koff]);
  }

  const float* a2base = attn + ((size_t)bh * S_ + i0) * S_;
  const int*   mbase  = mask + ((size_t)b  * S_ + i0) * S_;

  // ---- Phase A: logits = QK^T + attn2, masked (wave-private K staging) -------
  for (int it = 0; it < 16; ++it) {
    const int j0 = (it * 4 + wave) * 16;
    {
      const unsigned int l0 = lds_off(&kstf[wave][0][0]);
      const unsigned long long g0 =
          (unsigned long long)(uintptr_t)(k + (((size_t)bh) * S_ + j0) * D_);
#pragma unroll
      for (int u = 0; u < 8; ++u) {
        const unsigned int off = (unsigned int)(u * 32 + lane) * 16u;
        async_b128(l0 + off, g0 + off);
      }
      asm volatile("s_wait_asynccnt 0" ::: "memory");
    }

    const int koffB = hiH * 16;
    v16h B0 = cvt16f(&kstf[wave][nm][0  + koffB]);
    v16h B1 = cvt16f(&kstf[wave][nm][32 + koffB]);

    v8f acc = {0.f, 0.f, 0.f, 0.f, 0.f, 0.f, 0.f, 0.f};
    acc = __builtin_amdgcn_wmma_f32_16x16x32_f16(false, A0, false, B0, (short)0, acc, false, false);
    acc = __builtin_amdgcn_wmma_f32_16x16x32_f16(false, A1, false, B1, (short)0, acc, false, false);

#pragma unroll
    for (int r = 0; r < 8; ++r) {
      const int m   = r + hiH * 8;
      float a2  = a2base[(size_t)m * S_ + j0 + nm];
      int   msk = mbase[(size_t)m * S_ + j0 + nm];
      float val = acc[r] + a2;
      Lg[m][j0 + nm] = (msk == 0) ? -1.0e9f : val;
    }
  }
  __syncthreads();

  // ---- Phase B: row softmax, write attn --------------------------------------
  {
    const int row = tid >> 3;   // 16 rows x 8 threads
    const int sub = tid & 7;

    float pm = -3.0e38f;
    for (int j = sub; j < S_; j += 8) pm = fmaxf(pm, Lg[row][j]);
    red[row][sub] = pm;
    __syncthreads();
    if (sub == 0) {
      float m2 = red[row][0];
#pragma unroll
      for (int u = 1; u < 8; ++u) m2 = fmaxf(m2, red[row][u]);
      rowmax[row] = m2;
    }
    __syncthreads();

    const float rm = rowmax[row];
    float ps = 0.f;
    for (int j = sub; j < S_; j += 8) {
      float e = __expf(Lg[row][j] - rm);
      Lg[row][j] = e;
      ps += e;
    }
    red[row][sub] = ps;
    __syncthreads();
    if (sub == 0) {
      float s2 = 0.f;
#pragma unroll
      for (int u = 0; u < 8; ++u) s2 += red[row][u];
      rowsum[row] = s2;
    }
    __syncthreads();

    const float inv = 1.0f / rowsum[row];
    float* aout = attn + ((size_t)bh * S_ + i0 + row) * S_;
    for (int j = sub; j < S_; j += 8) {
      float p = Lg[row][j] * inv;
      Lg[row][j] = p;
      aout[j] = p;
    }
    __syncthreads();
  }

  // ---- Phase C: output = P (16x1024) * V (1024x64), wave w owns cols w*16.. --
  const int n0 = wave * 16;
  v8f oacc = {0.f, 0.f, 0.f, 0.f, 0.f, 0.f, 0.f, 0.f};
  for (int kk = 0; kk < S_; kk += 32) {
    {
      const int r     = tid >> 2;          // 0..31 rows of the V chunk
      const int dbase = (tid & 3) * 16;
      const float4* vp = (const float4*)(v + (((size_t)bh) * S_ + kk + r) * D_ + dbase);
#pragma unroll
      for (int u = 0; u < 4; ++u) {
        float4 f = vp[u];
        vt[dbase + 4 * u + 0][r] = (_Float16)f.x;
        vt[dbase + 4 * u + 1][r] = (_Float16)f.y;
        vt[dbase + 4 * u + 2][r] = (_Float16)f.z;
        vt[dbase + 4 * u + 3][r] = (_Float16)f.w;
      }
    }
    __syncthreads();

    v16h Ap;
    {
      const int koffA = hiH * 16;
#pragma unroll
      for (int e = 0; e < 8; ++e) Ap[e]     = (_Float16)Lg[nm][kk + koffA + e];
#pragma unroll
      for (int e = 0; e < 8; ++e) Ap[8 + e] = (_Float16)Lg[nm][kk + 8 + koffA + e];
    }
    v16h Bp = *(const v16h*)&vt[n0 + nm][hiH * 16];

    oacc = __builtin_amdgcn_wmma_f32_16x16x32_f16(false, Ap, false, Bp, (short)0, oacc, false, false);
    __syncthreads();
  }

  float* obase = out + ((size_t)bh * S_ + i0) * D_ + n0;
#pragma unroll
  for (int r = 0; r < 8; ++r) {
    const int m = r + hiH * 8;
    obase[(size_t)m * D_ + nm] = oacc[r];
  }
}

// -----------------------------------------------------------------------------
extern "C" void kernel_launch(void* const* d_in, const int* in_sizes, int n_in,
                              void* d_out, int out_size, void* d_ws, size_t ws_size,
                              hipStream_t stream) {
  const float* q    = (const float*)d_in[0];
  const float* k    = (const float*)d_in[1];
  const float* v    = (const float*)d_in[2];
  const int*   mask = (const int*)d_in[3];
  const float* rpr  = (const float*)d_in[4];

  float* out  = (float*)d_out;                                  // (B,H,S,Dv)
  float* attn = (float*)d_out + (size_t)B_ * H_ * S_ * D_;      // (B,H,S,S)

  hipLaunchKernelGGL(attn2_kernel, dim3(B_ * S_), dim3(32), 0, stream,
                     q, rpr, attn);
  hipLaunchKernelGGL(attn_fused_kernel, dim3(B_ * H_ * (S_ / 16)), dim3(128), 0, stream,
                     q, k, v, mask, attn, out);
}
